// KNNSoftmax_6562710028566
// MI455X (gfx1250) — compile-verified, matrix operand order checked by reading, not run on hardware
//
#include <hip/hip_runtime.h>
#include <hip/hip_bf16.h>
#include <stdint.h>

// ---------------------------------------------------------------------------
// Problem constants (reference: x[4096,1024] fp32 L2-normalized, t[4096] int)
// ---------------------------------------------------------------------------
#define NPTS 4096
#define DDIM 1024
#define ALPHA 30.0f
#define KSEL 16            // threshold = (KSEL+1)-th smallest off-diag dist

typedef __attribute__((ext_vector_type(16))) __bf16 v16bf;
typedef __attribute__((ext_vector_type(8)))  __bf16 v8bf;
typedef __attribute__((ext_vector_type(8)))  float  v8f;

// ---------------------------------------------------------------------------
// 1) fp32 -> bf16 hi/lo split (error-compensated bf16 GEMM inputs)
// ---------------------------------------------------------------------------
__global__ __launch_bounds__(256) void knn_split_bf16(const float* __restrict__ x,
                                                      __bf16* __restrict__ xhi,
                                                      __bf16* __restrict__ xlo,
                                                      int total) {
    int idx = blockIdx.x * blockDim.x + threadIdx.x;
    if (idx >= total) return;
    float v = x[idx];
    __bf16 h = (__bf16)v;
    float hf = (float)h;
    __bf16 l = (__bf16)(v - hf);
    xhi[idx] = h;
    xlo[idx] = l;
}

// ---------------------------------------------------------------------------
// 2) Gram matrix G = X Xt via v_wmma_f32_16x16x32_bf16, hi/lo compensated.
//    Workgroup = 256 threads = 8 waves arranged 4x2; each wave computes a
//    32x32 output block (2x2 WMMA tiles, 4 x v8f accumulators).
//    Block output tile = 128 rows x 64 cols.
//
//    Fragment layout per ISA 7.12.2 (16-bit A 16x32): lane m in 0..15 holds
//    row m, K = 0..7 and 16..23 (lanes 16..31 hold K=8..15 / 24..31), packed
//    2 bf16 per VGPR.  B(32x16) for G = X*Xt mirrors this with N=lane, so
//    both fragments are loaded with the identical per-lane pattern.
// ---------------------------------------------------------------------------
__device__ inline v16bf knn_load_frag(const __bf16* __restrict__ p) {
    // p points at (row, kc + kh*8); elements 0..7 = K+0..7, 8..15 = K+16..23
    v8bf a = *(const v8bf*)(p);
    v8bf b = *(const v8bf*)(p + 16);
    v16bf f;
#pragma unroll
    for (int i = 0; i < 8; ++i) { f[i] = a[i]; f[i + 8] = b[i]; }
    return f;
}

__global__ __launch_bounds__(256) void knn_gram_wmma(const __bf16* __restrict__ xhi,
                                                     const __bf16* __restrict__ xlo,
                                                     float* __restrict__ G) {
    const int lane    = threadIdx.x & 31;
    const int wave    = threadIdx.x >> 5;   // 0..7
    const int waveRow = wave & 3;           // 0..3 -> 4 * 32 = 128 rows
    const int waveCol = wave >> 2;          // 0..1 -> 2 * 32 =  64 cols
    const int rowBase = blockIdx.y * 128 + waveRow * 32;
    const int colBase = blockIdx.x * 64  + waveCol * 32;
    const int mIdx    = lane & 15;
    const int kh      = lane >> 4;          // which K half this lane holds

    v8f acc[2][2];
#pragma unroll
    for (int a = 0; a < 2; ++a)
#pragma unroll
        for (int b = 0; b < 2; ++b)
#pragma unroll
            for (int v = 0; v < 8; ++v) acc[a][b][v] = 0.0f;

    for (int kc = 0; kc < DDIM; kc += 32) {
        v16bf ahi[2], alo[2], bhi[2], blo[2];
#pragma unroll
        for (int t = 0; t < 2; ++t) {
            const size_t ra = (size_t)(rowBase + t * 16 + mIdx) * DDIM + kc + kh * 8;
            const size_t rb = (size_t)(colBase + t * 16 + mIdx) * DDIM + kc + kh * 8;
            ahi[t] = knn_load_frag(xhi + ra);
            alo[t] = knn_load_frag(xlo + ra);
            bhi[t] = knn_load_frag(xhi + rb);
            blo[t] = knn_load_frag(xlo + rb);
        }
#pragma unroll
        for (int mi = 0; mi < 2; ++mi) {
#pragma unroll
            for (int ni = 0; ni < 2; ++ni) {
                // hi*hi + hi*lo + lo*hi  (lo*lo term ~2^-32, dropped)
                acc[mi][ni] = __builtin_amdgcn_wmma_f32_16x16x32_bf16(
                    false, ahi[mi], false, bhi[ni], (short)0, acc[mi][ni], false, false);
                acc[mi][ni] = __builtin_amdgcn_wmma_f32_16x16x32_bf16(
                    false, ahi[mi], false, blo[ni], (short)0, acc[mi][ni], false, false);
                acc[mi][ni] = __builtin_amdgcn_wmma_f32_16x16x32_bf16(
                    false, alo[mi], false, bhi[ni], (short)0, acc[mi][ni], false, false);
            }
        }
    }

    // C/D layout: VGPR v, lanes 0..15 -> M=v, N=lane; lanes 16..31 -> M=v+8.
#pragma unroll
    for (int mi = 0; mi < 2; ++mi) {
#pragma unroll
        for (int ni = 0; ni < 2; ++ni) {
#pragma unroll
            for (int v = 0; v < 8; ++v) {
                int r = rowBase + mi * 16 + v + kh * 8;
                int c = colBase + ni * 16 + mIdx;
                G[(size_t)r * NPTS + c] = acc[mi][ni][v];
            }
        }
    }
}

// ---------------------------------------------------------------------------
// 3) sq[i] = G[i][i]
// ---------------------------------------------------------------------------
__global__ __launch_bounds__(256) void knn_diag(const float* __restrict__ G,
                                                float* __restrict__ sq) {
    int i = blockIdx.x * blockDim.x + threadIdx.x;
    if (i < NPTS) sq[i] = G[(size_t)i * NPTS + i];
}

// ---------------------------------------------------------------------------
// 4) zero the global accumulators (sum_loss, n_valid, acc_count)
// ---------------------------------------------------------------------------
__global__ void knn_zero(float* __restrict__ accum) {
    if (threadIdx.x < 3) accum[threadIdx.x] = 0.0f;
}

// ---------------------------------------------------------------------------
// 5) per-row: distances -> exact 17th-smallest threshold via bit-pattern
//    bisection over LDS -> masked exp sums -> atomically accumulate.
//    One 256-thread workgroup per row; the full 16KB row sits in LDS
//    (320KB/WGP on CDNA5 makes this cheap).
// ---------------------------------------------------------------------------
__global__ __launch_bounds__(256) void knn_row(const float* __restrict__ G,
                                               const float* __restrict__ sq,
                                               const int* __restrict__ targets,
                                               float* __restrict__ accum) {
    __shared__ float sdist[NPTS];
    __shared__ unsigned scount;
    __shared__ float s_pos, s_neg;
    __shared__ int s_first;
    __shared__ unsigned s_flags;

    const int i   = blockIdx.x;
    const int tid = threadIdx.x;
    const float sqi = sq[i];
    const float* g = G + (size_t)i * NPTS;

    for (int j = tid; j < NPTS; j += blockDim.x) {
        float d2 = sqi + sq[j] - 2.0f * g[j];
        float d  = sqrtf(fmaxf(d2, 1e-12f));
        sdist[j] = (j == i) ? 1e30f : d;   // diag excluded from selection
    }
    if (tid == 0) { s_pos = 0.0f; s_neg = 0.0f; s_first = 0x7fffffff; s_flags = 0u; }
    __syncthreads();

    // Exact k-th smallest: smallest uint t with |{j : bits(d_j) <= t}| >= K+1.
    // Non-negative IEEE floats are order-isomorphic to their bit patterns.
    unsigned lo = 0u, hi = 0x7f7fffffu;
    while (lo < hi) {
        unsigned mid = lo + ((hi - lo) >> 1);
        if (tid == 0) scount = 0u;
        __syncthreads();
        unsigned c = 0;
        for (int j = tid; j < NPTS; j += blockDim.x)
            c += (__float_as_uint(sdist[j]) <= mid) ? 1u : 0u;
        atomicAdd(&scount, c);
        __syncthreads();
        unsigned total = scount;
        if (total >= (KSEL + 1)) hi = mid; else lo = mid + 1;
        __syncthreads();
    }
    const float thr = __uint_as_float(lo);

    const int myt = targets[i];
    float psum = 0.0f, nsum = 0.0f;
    unsigned flags = 0u;            // bit0 anyPos, bit1 anyNeg, bit2 anyPosBelow
    int firstPos = 0x7fffffff;
    for (int j = tid; j < NPTS; j += blockDim.x) {
        if (j == i) continue;
        const float d = sdist[j];
        const bool same  = (targets[j] == myt);
        const bool below = (d < thr);
        const float logit = expf(ALPHA * (1.0f - d));
        if (same) {
            flags |= 1u;
            if (j < firstPos) firstPos = j;
            if (below) { flags |= 4u; psum += logit; }
        } else {
            flags |= 2u;
            if (below) nsum += logit;
        }
    }
    atomicAdd(&s_pos, psum);
    atomicAdd(&s_neg, nsum);
    atomicOr(&s_flags, flags);
    atomicMin(&s_first, firstPos);
    __syncthreads();

    if (tid == 0) {
        unsigned f = s_flags;
        bool anyPos = (f & 1u), anyNeg = (f & 2u), anyPB = (f & 4u);
        if (anyPos && anyNeg) {
            float pos = anyPB ? s_pos
                              : expf(ALPHA * (1.0f - sdist[s_first]));
            float li = -logf(pos / (pos + s_neg));
            atomicAdd(&accum[0], li);           // sum of valid losses
            atomicAdd(&accum[1], 1.0f);         // n_valid
            if (li < 0.6f) atomicAdd(&accum[2], 1.0f);
        }
    }
}

// ---------------------------------------------------------------------------
// 6) finalize: (loss, accuracy, 0, 0)
// ---------------------------------------------------------------------------
__global__ void knn_finalize(const float* __restrict__ accum,
                             float* __restrict__ out) {
    if (threadIdx.x == 0) {
        out[0] = accum[0] / accum[1];
        out[1] = accum[2] / (float)NPTS;
        out[2] = 0.0f;
        out[3] = 0.0f;
    }
}

// ---------------------------------------------------------------------------
// launch
// ---------------------------------------------------------------------------
extern "C" void kernel_launch(void* const* d_in, const int* in_sizes, int n_in,
                              void* d_out, int out_size, void* d_ws, size_t ws_size,
                              hipStream_t stream) {
    (void)in_sizes; (void)n_in; (void)out_size; (void)ws_size;
    const float* x       = (const float*)d_in[0];
    const int*   targets = (const int*)d_in[1];
    float* out = (float*)d_out;

    // workspace layout
    char* ws = (char*)d_ws;
    __bf16* xhi = (__bf16*)(ws);                                  //  8 MB
    __bf16* xlo = (__bf16*)(ws + (size_t)NPTS * DDIM * 2);        //  8 MB
    float*  G   = (float*) (ws + (size_t)NPTS * DDIM * 4);        // 64 MB
    float*  sq  = (float*) (ws + (size_t)NPTS * DDIM * 4
                               + (size_t)NPTS * NPTS * 4);        // 16 KB
    float*  accum = sq + NPTS;                                    // 3 floats

    const int total = NPTS * DDIM;
    knn_split_bf16<<<(total + 255) / 256, 256, 0, stream>>>(x, xhi, xlo, total);

    dim3 ggrid(NPTS / 64, NPTS / 128);       // 64 x 32 workgroups
    knn_gram_wmma<<<ggrid, 256, 0, stream>>>(xhi, xlo, G);

    knn_diag<<<(NPTS + 255) / 256, 256, 0, stream>>>(G, sq);
    knn_zero<<<1, 32, 0, stream>>>(accum);
    knn_row<<<NPTS, 256, 0, stream>>>(G, sq, targets, accum);
    knn_finalize<<<1, 32, 0, stream>>>(accum, out);
}
